// PointNetFeaturePropagation_35296041239307
// MI455X (gfx1250) — matrix operand order repaired and checked by reading, not compiled, not run on hardware
//
#include <hip/hip_runtime.h>

// ---------------------------------------------------------------------------
// PointNet Feature Propagation for MI455X (gfx1250, wave32, WMMA)
//   stage 1: 3-NN (xyz2 tile in LDS) -> idx/weights
//   stage 2: build X = [interp(features2) | features1] as bf16 [65536 x 384]
//   stage 3: GEMM1 via v_wmma_f32_16x16x32_bf16 (64x64 block tiles,
//            weight panel staged with global_load_async_to_lds_b128)
//   stage 4: BN batch-stats reduce (atomics) + apply + ReLU -> bf16
//   stage 5: GEMM2 via WMMA -> O_pre fp32, BN stats, apply -> d_out fp32
// ---------------------------------------------------------------------------

typedef __attribute__((ext_vector_type(16))) __bf16 v16bf;
typedef __attribute__((ext_vector_type(8)))  float  v8f;

union BfPack { uint4 u[2]; v16bf v; };

constexpr int   BB    = 8;
constexpr int   NPTS  = 8192;
constexpr int   SPTS  = 2048;
constexpr int   C1    = 128;
constexpr int   C2    = 256;
constexpr int   CIN   = 384;   // C1 + C2
constexpr int   H1    = 256;
constexpr int   H2    = 128;
constexpr int   MROWS = BB * NPTS;  // 65536
constexpr float BN_EPS = 1e-5f;

// async global->LDS 16B copy (CDNA5, ASYNCcnt-tracked)
__device__ __forceinline__ void async_g2l_b128(unsigned lds_off, const void* gaddr) {
    asm volatile("global_load_async_to_lds_b128 %0, %1, off"
                 :: "v"(lds_off), "v"(gaddr)
                 : "memory");
}
__device__ __forceinline__ void wait_asynccnt0() {
    asm volatile("s_wait_asynccnt 0x0" ::: "memory");
}

// ---------------------------------------------------------------- utilities
__global__ void zero_kernel(float* __restrict__ p, int n) {
    int i = blockIdx.x * blockDim.x + threadIdx.x;
    if (i < n) p[i] = 0.0f;
}

__global__ void cvt_bf16_kernel(const float* __restrict__ src,
                                __bf16* __restrict__ dst, int n) {
    int i = blockIdx.x * blockDim.x + threadIdx.x;
    if (i < n) dst[i] = (__bf16)src[i];
}

// ------------------------------------------------------------- 3-NN kernel
__global__ __launch_bounds__(256)
void knn3_kernel(const float* __restrict__ xyz1, const float* __restrict__ xyz2,
                 int* __restrict__ idx3, float* __restrict__ w3) {
    __shared__ float sx[SPTS], sy[SPTS], sz[SPTS];
    const int batch = blockIdx.y;
    const float* p2 = xyz2 + (size_t)batch * SPTS * 3;
    for (int i = threadIdx.x; i < SPTS; i += blockDim.x) {
        sx[i] = p2[i * 3 + 0];
        sy[i] = p2[i * 3 + 1];
        sz[i] = p2[i * 3 + 2];
    }
    __syncthreads();

    const int n = blockIdx.x * blockDim.x + threadIdx.x;
    const size_t row = (size_t)batch * NPTS + n;
    const float px = xyz1[row * 3 + 0];
    const float py = xyz1[row * 3 + 1];
    const float pz = xyz1[row * 3 + 2];

    float b0 = 3.4e38f, b1v = 3.4e38f, b2v = 3.4e38f;
    int   i0 = 0, i1 = 0, i2 = 0;
    for (int s = 0; s < SPTS; ++s) {
        const float dx = px - sx[s], dy = py - sy[s], dz = pz - sz[s];
        const float d2 = dx * dx + dy * dy + dz * dz;
        if (d2 < b0)       { b2v = b1v; i2 = i1; b1v = b0; i1 = i0; b0 = d2; i0 = s; }
        else if (d2 < b1v) { b2v = b1v; i2 = i1; b1v = d2; i1 = s; }
        else if (d2 < b2v) { b2v = d2; i2 = s; }
    }
    const float w0 = 1.0f / (sqrtf(fmaxf(b0,  0.0f)) + 1e-10f);
    const float w1 = 1.0f / (sqrtf(fmaxf(b1v, 0.0f)) + 1e-10f);
    const float w2 = 1.0f / (sqrtf(fmaxf(b2v, 0.0f)) + 1e-10f);
    const float wi = 1.0f / (w0 + w1 + w2);
    idx3[row * 3 + 0] = i0;  w3[row * 3 + 0] = w0 * wi;
    idx3[row * 3 + 1] = i1;  w3[row * 3 + 1] = w1 * wi;
    idx3[row * 3 + 2] = i2;  w3[row * 3 + 2] = w2 * wi;
}

// ------------------------------------------------- build X (bf16, coalesced)
__global__ __launch_bounds__(128)
void buildx_kernel(const float* __restrict__ f1, const float* __restrict__ f2,
                   const int* __restrict__ idx3, const float* __restrict__ w3,
                   __bf16* __restrict__ X) {
    const size_t row   = blockIdx.x;
    const int    batch = (int)(row / NPTS);
    const int i0 = idx3[row * 3 + 0], i1 = idx3[row * 3 + 1], i2 = idx3[row * 3 + 2];
    const float w0 = w3[row * 3 + 0], w1 = w3[row * 3 + 1], w2 = w3[row * 3 + 2];
    const float* r0 = f2 + ((size_t)batch * SPTS + i0) * C2;
    const float* r1 = f2 + ((size_t)batch * SPTS + i1) * C2;
    const float* r2 = f2 + ((size_t)batch * SPTS + i2) * C2;
    __bf16* xr = X + row * CIN;
    for (int c = threadIdx.x; c < C2; c += blockDim.x)
        xr[c] = (__bf16)(w0 * r0[c] + w1 * r1[c] + w2 * r2[c]);
    const float* f1r = f1 + row * C1;
    for (int c = threadIdx.x; c < C1; c += blockDim.x)
        xr[C2 + c] = (__bf16)f1r[c];
}

// ------------------------------------------------------------- WMMA GEMM
// out[m,o] = sum_k A[m,k] * W[o,k] + bias[o]
// block = 4 waves -> 64(M) x 64(O) tile. Each wave owns a 16x64 strip:
// one A fragment feeds 4 WMMAs (4 O-tiles), quadrupling math per A byte.
// The 64-column weight panel (KDIM x 64 bf16) is staged into LDS with
// CDNA5 async global->LDS b128 transfers (ASYNCcnt), then read as B
// fragments via ds_load_b128.
template <int KDIM, int ODIM>
__global__ __launch_bounds__(128)
void gemm_wmma_kernel(const __bf16* __restrict__ A, const __bf16* __restrict__ W,
                      const float* __restrict__ bias, float* __restrict__ out) {
    __shared__ __align__(16) __bf16 Wlds[KDIM * 64];   // <= 48 KB

    const int tid  = threadIdx.x;
    const int wave = tid >> 5;
    const int lane = tid & 31;
    const int ocol0 = blockIdx.y * 64;

    // 64 consecutive rows of W (row-major [ODIM][KDIM]) form the contiguous
    // B panel: async-copy global -> LDS in 16B chunks, no VGPR staging.
    {
        const char* src = reinterpret_cast<const char*>(W + (size_t)ocol0 * KDIM);
        const unsigned lds_base = (unsigned)(uintptr_t)(void*)Wlds;
        constexpr int CHUNKS = KDIM * 64 * 2 / 16;     // 16B chunks
        for (int i = tid; i < CHUNKS; i += 128)
            async_g2l_b128(lds_base + i * 16, src + (size_t)i * 16);
        wait_asynccnt0();
    }
    __syncthreads();

    const int row0 = (blockIdx.x * 4 + wave) * 16;
    const int lm = lane & 15;     // M row (A) / N col (B,C,D)
    const int lh = lane >> 4;     // K-half selector

    // A lane layout (16-bit A 16x32): lane<16 -> K {kb..kb+7, kb+16..kb+23},
    // lane>=16 -> +8 in each half. Two 16B loads per step.
    const __bf16* arow = A + (size_t)(row0 + lm) * KDIM + lh * 8;
    // B lane layout (16-bit B 32x16): lane<16 -> col=lane, K {kb..kb+7, kb+16..kb+23};
    // lane>=16 -> col=lane-16, +8 in each half. Two 16B loads per step.
    const __bf16* brow0 = Wlds + lm * KDIM + lh * 8;

    v8f acc[4];
#pragma unroll
    for (int j = 0; j < 4; ++j) {
        const float bv = bias[ocol0 + j * 16 + lm];
#pragma unroll
        for (int i = 0; i < 8; ++i) acc[j][i] = bv;
    }

#pragma unroll
    for (int kb = 0; kb < KDIM; kb += 32) {
        BfPack pa;
        pa.u[0] = *reinterpret_cast<const uint4*>(arow + kb);
        pa.u[1] = *reinterpret_cast<const uint4*>(arow + kb + 16);
        __builtin_prefetch(arow + kb + 64, 0, 1);      // global_prefetch_b8
#pragma unroll
        for (int j = 0; j < 4; ++j) {
            BfPack pb;
            const __bf16* bj = brow0 + (size_t)j * 16 * KDIM + kb;
            pb.u[0] = *reinterpret_cast<const uint4*>(bj);
            pb.u[1] = *reinterpret_cast<const uint4*>(bj + 16);
            acc[j] = __builtin_amdgcn_wmma_f32_16x16x32_bf16(
                /*neg_a=*/false, pa.v, /*neg_b=*/false, pb.v,
                /*c_mod=*/(short)0, acc[j], /*reuse_a=*/false, /*reuse_b=*/false);
        }
    }

    // C/D layout: VGPR r -> M = lh*8 + r, N = lm
#pragma unroll
    for (int j = 0; j < 4; ++j) {
        size_t base = (size_t)(row0 + lh * 8) * ODIM + ocol0 + j * 16 + lm;
#pragma unroll
        for (int r = 0; r < 8; ++r) out[base + (size_t)r * ODIM] = acc[j][r];
    }
}

// ---------------------------------------------------------- BN batch stats
template <int ODIM>
__global__ void bn_stats_kernel(const float* __restrict__ H, float* __restrict__ stats) {
    const int c = threadIdx.x;            // blockDim.x == ODIM
    float s = 0.0f, s2 = 0.0f;
    for (int r = blockIdx.x; r < MROWS; r += gridDim.x) {
        const float v = H[(size_t)r * ODIM + c];
        s += v; s2 += v * v;
    }
    atomicAdd(&stats[c], s);
    atomicAdd(&stats[ODIM + c], s2);
}

template <int ODIM>
__global__ void bn_apply_bf16_kernel(const float* __restrict__ H,
                                     const float* __restrict__ stats,
                                     const float* __restrict__ g,
                                     const float* __restrict__ be,
                                     __bf16* __restrict__ out) {
    const size_t i = (size_t)blockIdx.x * blockDim.x + threadIdx.x;
    if (i >= (size_t)MROWS * ODIM) return;
    const int c = (int)(i & (ODIM - 1));
    const float mean = stats[c] * (1.0f / MROWS);
    const float var  = stats[ODIM + c] * (1.0f / MROWS) - mean * mean;
    const float sc   = rsqrtf(var + BN_EPS) * g[c];
    const float y    = (H[i] - mean) * sc + be[c];
    out[i] = (__bf16)fmaxf(y, 0.0f);
}

template <int ODIM>
__global__ void bn_apply_f32_kernel(const float* __restrict__ H,
                                    const float* __restrict__ stats,
                                    const float* __restrict__ g,
                                    const float* __restrict__ be,
                                    float* __restrict__ out) {
    const size_t i = (size_t)blockIdx.x * blockDim.x + threadIdx.x;
    if (i >= (size_t)MROWS * ODIM) return;
    const int c = (int)(i & (ODIM - 1));
    const float mean = stats[c] * (1.0f / MROWS);
    const float var  = stats[ODIM + c] * (1.0f / MROWS) - mean * mean;
    const float sc   = rsqrtf(var + BN_EPS) * g[c];
    const float y    = (H[i] - mean) * sc + be[c];
    out[i] = fmaxf(y, 0.0f);
}

// ---------------------------------------------------------------- launcher
extern "C" void kernel_launch(void* const* d_in, const int* in_sizes, int n_in,
                              void* d_out, int out_size, void* d_ws, size_t ws_size,
                              hipStream_t stream) {
    const float* xyz1 = (const float*)d_in[0];
    const float* xyz2 = (const float*)d_in[1];
    const float* f1   = (const float*)d_in[2];
    const float* f2   = (const float*)d_in[3];
    const float* W1   = (const float*)d_in[4];
    const float* b1   = (const float*)d_in[5];
    const float* g1   = (const float*)d_in[6];
    const float* be1  = (const float*)d_in[7];
    const float* W2   = (const float*)d_in[8];
    const float* b2   = (const float*)d_in[9];
    const float* g2   = (const float*)d_in[10];
    const float* be2  = (const float*)d_in[11];
    float* out = (float*)d_out;

    // ---- workspace layout ----
    char*  w   = (char*)d_ws;
    size_t off = 0;
    auto take = [&](size_t bytes) { char* p = w + off; off += (bytes + 255) & ~(size_t)255; return p; };

    __bf16* X      = (__bf16*)take((size_t)MROWS * CIN * 2);   // 50.3 MB
    float*  H1pre  = (float*) take((size_t)MROWS * H1 * 4);    // 67.1 MB
    __bf16* H1b    = (__bf16*)take((size_t)MROWS * H1 * 2);    // 33.6 MB
    float*  O2pre  = (float*) take((size_t)MROWS * H2 * 4);    // 33.6 MB
    int*    idx3   = (int*)   take((size_t)MROWS * 3 * 4);
    float*  w3     = (float*) take((size_t)MROWS * 3 * 4);
    __bf16* W1b    = (__bf16*)take((size_t)H1 * CIN * 2);
    __bf16* W2b    = (__bf16*)take((size_t)H2 * H1 * 2);
    float*  stats1 = (float*) take(2 * H1 * 4);
    float*  stats2 = (float*) take(2 * H2 * 4);
    (void)ws_size; (void)in_sizes; (void)n_in; (void)out_size;

    // 0) zero BN stat accumulators
    zero_kernel<<<1, 256, 0, stream>>>(stats1, 2 * H1);
    zero_kernel<<<1, 256, 0, stream>>>(stats2, 2 * H2);

    // 1) weights -> bf16
    cvt_bf16_kernel<<<(H1 * CIN + 255) / 256, 256, 0, stream>>>(W1, W1b, H1 * CIN);
    cvt_bf16_kernel<<<(H2 * H1  + 255) / 256, 256, 0, stream>>>(W2, W2b, H2 * H1);

    // 2) 3-NN with inverse-distance weights
    knn3_kernel<<<dim3(NPTS / 256, BB), 256, 0, stream>>>(xyz1, xyz2, idx3, w3);

    // 3) X = [interp(features2) | features1] in bf16
    buildx_kernel<<<MROWS, 128, 0, stream>>>(f1, f2, idx3, w3, X);

    // 4) GEMM1 (65536x384 @ 384x256) + bias -> H1pre
    gemm_wmma_kernel<CIN, H1><<<dim3(MROWS / 64, H1 / 64), 128, 0, stream>>>(X, W1b, b1, H1pre);

    // 5) BN1 stats + apply(ReLU) -> bf16
    bn_stats_kernel<H1><<<512, H1, 0, stream>>>(H1pre, stats1);
    bn_apply_bf16_kernel<H1><<<(MROWS * H1 + 255) / 256, 256, 0, stream>>>(H1pre, stats1, g1, be1, H1b);

    // 6) GEMM2 (65536x256 @ 256x128) + bias -> O2pre
    gemm_wmma_kernel<H1, H2><<<dim3(MROWS / 64, H2 / 64), 128, 0, stream>>>(H1b, W2b, b2, O2pre);

    // 7) BN2 stats + apply(ReLU) -> d_out (fp32)
    bn_stats_kernel<H2><<<512, H2, 0, stream>>>(O2pre, stats2);
    bn_apply_f32_kernel<H2><<<(MROWS * H2 + 255) / 256, 256, 0, stream>>>(O2pre, stats2, g2, be2, out);
}
